// ATSS_86878598464115
// MI455X (gfx1250) — matrix operand order repaired and checked by reading, not compiled, not run on hardware
//
#include <hip/hip_runtime.h>
#include <math.h>

// ---------------------------------------------------------------- types
typedef __attribute__((ext_vector_type(16))) __bf16 v16bf;
typedef __attribute__((ext_vector_type(8)))  float  v8f;
typedef __attribute__((ext_vector_type(4)))  int    v4i_;

union FragB { v16bf v; uint4 q[2]; };

#define CIN        256
#define KT_TOTAL   16            // 256/16 output-channel tiles
#define TAPS       9
#define CBLKS      8             // 256/32 K blocks
#define FRAG_PER_CONV (KT_TOTAL*TAPS*CBLKS*32*16)   // 589824 bf16
#define HEAD_KT    6
#define HEAD_FRAG  (HEAD_KT*CBLKS*32*16)            // 24576 bf16
#define MAXC       4096
#define NIMG       4
#define NMS_KEEP   100
#define ROWS       4             // output rows per conv block
#define SCALE_CLAMP 4.135166556742356f
#define IMG_SIZE   800.0f

// gfx1250 async global->LDS copy path (ASYNCcnt), guarded so compile never breaks.
#if defined(__AMDGCN__) && __has_builtin(__builtin_amdgcn_global_load_async_to_lds_b128) && \
    __has_builtin(__builtin_amdgcn_s_wait_asynccnt)
#define HAVE_ASYNC_LDS 1
#else
#define HAVE_ASYNC_LDS 0
#endif

__device__ inline void copy_b128_g2l(const void* gsrc, void* ldst) {
#if HAVE_ASYNC_LDS
  __builtin_amdgcn_global_load_async_to_lds_b128(
      (__attribute__((address_space(1))) v4i_*)gsrc,
      (__attribute__((address_space(3))) v4i_*)ldst, 0, 0);
#else
  *(uint4*)ldst = *(const uint4*)gsrc;
#endif
}
__device__ inline void async_fence() {
#if HAVE_ASYNC_LDS
  __builtin_amdgcn_s_wait_asynccnt(0);
#endif
}

__device__ inline unsigned short f2bf(float f) {
  union { float f; unsigned u; } x; x.f = f;
  unsigned u = x.u;
  u += 0x7FFFu + ((u >> 16) & 1u);   // round-to-nearest-even
  return (unsigned short)(u >> 16);
}
__device__ inline float sigmoidf_(float x) { return 1.0f / (1.0f + __expf(-x)); }

// ------------------------------------------- weight fragment preparation
// Fragment layout [kt][tap][cb][lane][16], lane half h holding
// channels cb*32+16h+j (contiguous k order, same as the B operand).
__global__ void prep_tower_w(const float* __restrict__ Wt, unsigned short* __restrict__ F) {
  int i = blockIdx.x * blockDim.x + threadIdx.x;
  if (i >= FRAG_PER_CONV) return;
  int j = i % 16;  int t = i / 16;
  int lane = t % 32; t /= 32;
  int cb   = t % 8;  t /= 8;
  int tap  = t % 9;  t /= 9;
  int kt   = t;
  int row = lane & 15;
  int k   = kt * 16 + row;
  int clocal = (lane >> 4) * 16 + j;
  int c = cb * 32 + clocal;
  int r = tap / 3, s = tap - 3 * r;
  F[i] = f2bf(Wt[(((size_t)k * CIN + c) * 3 + r) * 3 + s]);
}

__global__ void prep_head_w(const float* __restrict__ wcls, const float* __restrict__ wbox,
                            const float* __restrict__ wctr, unsigned short* __restrict__ F) {
  int i = blockIdx.x * blockDim.x + threadIdx.x;
  if (i >= HEAD_FRAG) return;
  int j = i % 16;  int t = i / 16;
  int lane = t % 32; t /= 32;
  int cb   = t % 8;  t /= 8;
  int kt   = t;
  int row = lane & 15;
  int k   = kt * 16 + row;
  int clocal = (lane >> 4) * 16 + j;
  int c = cb * 32 + clocal;
  float v = 0.f;
  if (k < 80)       v = wcls[(size_t)k * CIN + c];
  else if (k < 84)  v = wbox[(size_t)(k - 80) * CIN + c];
  else if (k == 84) v = wctr[c];
  F[i] = f2bf(v);
}

__global__ void prep_bias96(const float* __restrict__ bcls, const float* __restrict__ bbox_b,
                            const float* __restrict__ bctr, float* __restrict__ B) {
  int i = threadIdx.x;
  if (i >= 96) return;
  float v = 0.f;
  if (i < 80)       v = bcls[i];
  else if (i < 84)  v = bbox_b[i - 80];
  else if (i == 84) v = bctr[0];
  B[i] = v;
}

__global__ void zero_cnt(int* cnt) { if (threadIdx.x < NIMG) cnt[threadIdx.x] = 0; }

// -------------------------- f32 NCHW -> bf16 NHWC conversion (per level)
__global__ void cvt_to_nhwc_bf16(const float* __restrict__ X, unsigned short* __restrict__ Y,
                                 int H, int W) {
  int total = NIMG * CIN * H * W;
  int gid = blockIdx.x * blockDim.x + threadIdx.x;
  if (gid >= total) return;
  int c = gid % CIN; int t = gid / CIN;
  int w = t % W; t /= W;
  int h = t % H; int n = t / H;
  Y[gid] = f2bf(X[(((size_t)n * CIN + c) * H + h) * W + w]);   // gid == ((n*H+h)*W+w)*CIN+c
}

// ------------------------------------------------------- 3x3 conv (WMMA)
// bf16 NHWC in/out. Block = 512 threads = 16 waves; tile = ROWS x 16 positions,
// all 256 output channels. Wave kt: A fragment loaded once per (tap, cb), four
// B fragments batched into distinct registers, then 4 back-to-back WMMAs.
__global__ __launch_bounds__(512)
void conv3x3_wmma(const unsigned short* __restrict__ X, const unsigned short* __restrict__ WF,
                  const float* __restrict__ bias, unsigned short* __restrict__ Y,
                  int H, int W) {
  const int CSTR = 264;                       // padded channel stride (ushorts)
  __shared__ __align__(16) unsigned short sX[(ROWS + 2) * 18 * CSTR];   // 55.7 KB

  int wt = (W + 15) >> 4;
  int h0 = (blockIdx.x / wt) * ROWS;
  int w0 = (blockIdx.x % wt) << 4;
  int n  = blockIdx.y;

  // Stage X[n, h0-1..h0+ROWS, w0-1..w0+16, :] into LDS in 16B channel chunks.
  for (int idx = threadIdx.x; idx < (ROWS + 2) * 18 * 32; idx += blockDim.x) {
    int cc = idx & 31; int t = idx >> 5;
    int wl = t % 18;   int dh = t / 18;
    int hh = h0 + dh - 1, ww = w0 + wl - 1;
    unsigned short* dst = &sX[(dh * 18 + wl) * CSTR + cc * 8];
    if (hh >= 0 && hh < H && ww >= 0 && ww < W) {
      const unsigned short* src = &X[(((size_t)n * H + hh) * W + ww) * CIN + cc * 8];
      copy_b128_g2l((const void*)src, (void*)dst);
    } else {
      uint4 z = {0u, 0u, 0u, 0u};
      *(uint4*)dst = z;
    }
  }
  async_fence();
  __syncthreads();

  int wave  = threadIdx.x >> 5;      // 0..15 = output-channel tile
  int lane  = threadIdx.x & 31;
  int col   = lane & 15;             // B-matrix column (w position)
  int khalf = (lane >> 4) << 4;      // lanes 16-31 hold K = 16..31

  v8f acc0 = {}, acc1 = {}, acc2 = {}, acc3 = {};

  for (int tap = 0; tap < TAPS; ++tap) {
    int r = tap / 3, s = tap - 3 * r;
    if (tap + 1 < TAPS)              // keep next tap's weights hot (global_prefetch_b8)
      __builtin_prefetch((const void*)((const uint4*)WF +
          ((((size_t)wave * TAPS + tap + 1) * CBLKS) * 32 + lane) * 2), 0, 1);
#pragma unroll
    for (int cb = 0; cb < CBLKS; ++cb) {
      FragB a;
      const uint4* wp = (const uint4*)WF +
          ((((size_t)wave * TAPS + tap) * CBLKS + cb) * 32 + lane) * 2;
      a.q[0] = wp[0]; a.q[1] = wp[1];
      // Batch all four B fragments into distinct registers so the ds_loads
      // issue together and the 4 WMMAs run back-to-back after one wait.
      FragB b0, b1, b2, b3;
      const unsigned short* base = &sX[(r * 18 + col + s) * CSTR + cb * 32 + khalf];
      const uint4* bp0 = (const uint4*)(base + 0 * 18 * CSTR);
      const uint4* bp1 = (const uint4*)(base + 1 * 18 * CSTR);
      const uint4* bp2 = (const uint4*)(base + 2 * 18 * CSTR);
      const uint4* bp3 = (const uint4*)(base + 3 * 18 * CSTR);
      b0.q[0] = bp0[0]; b0.q[1] = bp0[1];
      b1.q[0] = bp1[0]; b1.q[1] = bp1[1];
      b2.q[0] = bp2[0]; b2.q[1] = bp2[1];
      b3.q[0] = bp3[0]; b3.q[1] = bp3[1];
      acc0 = __builtin_amdgcn_wmma_f32_16x16x32_bf16(false, a.v, false, b0.v, (short)0, acc0, false, false);
      acc1 = __builtin_amdgcn_wmma_f32_16x16x32_bf16(false, a.v, false, b1.v, (short)0, acc1, false, false);
      acc2 = __builtin_amdgcn_wmma_f32_16x16x32_bf16(false, a.v, false, b2.v, (short)0, acc2, false, false);
      acc3 = __builtin_amdgcn_wmma_f32_16x16x32_bf16(false, a.v, false, b3.v, (short)0, acc3, false, false);
    }
  }

  int wglob = w0 + col;
  if (wglob < W) {
    int k0 = (wave << 4) + ((lane >> 4) << 3);   // 8 consecutive out channels
    float b[8];
#pragma unroll
    for (int j = 0; j < 8; ++j) b[j] = bias[k0 + j];
    v8f accs[ROWS] = {acc0, acc1, acc2, acc3};
#pragma unroll
    for (int rr = 0; rr < ROWS; ++rr) {
      int hh = h0 + rr;
      if (hh < H) {
        union { unsigned short h[8]; uint4 q; } o;
#pragma unroll
        for (int j = 0; j < 8; ++j) {
          float v = accs[rr][j] + b[j];
          o.h[j] = f2bf(fmaxf(v, 0.f));          // tower convs are all ReLU
        }
        *(uint4*)&Y[(((size_t)n * H + hh) * W + wglob) * CIN + k0] = o.q;  // packed b128
      }
    }
  }
}

// ----------------------------------------------- 1x1 head convs (WMMA)
// 96 padded output channels: kt 0..4 = 80 cls logits (cls tower),
// kt 5 = {4 box deltas, 1 centerness, pad} (bbox tower). bf16 NHWC input.
__global__ __launch_bounds__(192)
void head_wmma(const unsigned short* __restrict__ clsF, const unsigned short* __restrict__ boxF,
               const unsigned short* __restrict__ WF, const float* __restrict__ bias96,
               float* __restrict__ HO, int HW) {
  const int CSTR = 264;
  __shared__ __align__(16) unsigned short sA[2 * 16 * CSTR];   // 16.9 KB

  int pos0 = blockIdx.x << 4;
  int n    = blockIdx.y;

  for (int idx = threadIdx.x; idx < 2 * 16 * 32; idx += blockDim.x) {
    int cc = idx & 31; int t = idx >> 5;
    int p = t & 15;    int which = t >> 4;
    int pos = pos0 + p;
    unsigned short* dst = &sA[(which * 16 + p) * CSTR + cc * 8];
    if (pos < HW) {
      const unsigned short* srcb = which ? boxF : clsF;
      copy_b128_g2l((const void*)&srcb[((size_t)n * HW + pos) * CIN + cc * 8], (void*)dst);
    } else {
      uint4 z = {0u, 0u, 0u, 0u};
      *(uint4*)dst = z;
    }
  }
  async_fence();
  __syncthreads();

  int wave  = threadIdx.x >> 5;     // 0..5
  int lane  = threadIdx.x & 31;
  int col   = lane & 15;
  int khalf = (lane >> 4) << 4;
  int srcsel = (wave < 5) ? 0 : 1;

  v8f acc = {};
#pragma unroll
  for (int cb = 0; cb < CBLKS; ++cb) {
    FragB a, b;
    const uint4* wp = (const uint4*)WF + (((size_t)wave * CBLKS + cb) * 32 + lane) * 2;
    a.q[0] = wp[0]; a.q[1] = wp[1];
    const uint4* bp = (const uint4*)(&sA[(srcsel * 16 + col) * CSTR + cb * 32 + khalf]);
    b.q[0] = bp[0]; b.q[1] = bp[1];
    acc = __builtin_amdgcn_wmma_f32_16x16x32_bf16(
        false, a.v, false, b.v, (short)0, acc, false, false);
  }

  int pos = pos0 + col;
  if (pos < HW) {
    int k0 = (wave << 4) + ((lane >> 4) << 3);
    float* op = &HO[((size_t)n * HW + pos) * 96 + k0];
#pragma unroll
    for (int j = 0; j < 8; ++j)
      op[j] = acc[j] + bias96[k0 + j];
  }
}

// --------------------------------- score/threshold/decode + compaction
__global__ void score_decode(const float* __restrict__ HO, const float* __restrict__ scales,
                             int lidx, int Hh, int Ww, float strideL,
                             float* __restrict__ cand, int* __restrict__ cnt) {
  int HW  = Hh * Ww;
  int gid = blockIdx.x * blockDim.x + threadIdx.x;
  if (gid >= NIMG * HW) return;
  int n = gid / HW, pos = gid % HW;
  const float* hp = HO + ((size_t)n * HW + pos) * 96;

  float sc  = scales[lidx];
  float ctr = sigmoidf_(hp[84]);
  int   hh  = pos / Ww, ww = pos % Ww;
  float asz = 8.f * strideL;
  float acx = (ww + 0.5f) * strideL, acy = (hh + 0.5f) * strideL;
  float dx  = hp[80] * sc * 0.1f,  dy  = hp[81] * sc * 0.1f;
  float dw  = fminf(hp[82] * sc * 0.2f, SCALE_CLAMP);
  float dh2 = fminf(hp[83] * sc * 0.2f, SCALE_CLAMP);
  float pcx = dx * asz + acx, pcy = dy * asz + acy;
  float pw  = __expf(dw) * asz, ph = __expf(dh2) * asz;
  float x1 = fminf(fmaxf(pcx - 0.5f * pw, 0.f), IMG_SIZE);
  float y1 = fminf(fmaxf(pcy - 0.5f * ph, 0.f), IMG_SIZE);
  float x2 = fminf(fmaxf(pcx + 0.5f * pw, 0.f), IMG_SIZE);
  float y2 = fminf(fmaxf(pcy + 0.5f * ph, 0.f), IMG_SIZE);

  for (int cl = 0; cl < 80; ++cl) {
    float s = sigmoidf_(hp[cl]) * ctr;
    if (s > 0.05f) {
      int slot = atomicAdd(&cnt[n], 1);
      if (slot < MAXC) {
        float* cp = cand + ((size_t)n * MAXC + slot) * 6;
        cp[0] = x1; cp[1] = y1; cp[2] = x2; cp[3] = y2;
        cp[4] = sqrtf(s); cp[5] = (float)cl;
      }
    }
  }
}

// --------------------------------------------- class-aware greedy NMS
__global__ __launch_bounds__(256)
void nms_kernel(const float* __restrict__ cand, const int* __restrict__ cnt,
                float* __restrict__ out) {
  __shared__ float ssc[MAXC];
  __shared__ float rv[256]; __shared__ int ri[256];
  __shared__ float wb[4];
  int n = blockIdx.x, tid = threadIdx.x;
  int M = cnt[n]; if (M > MAXC) M = MAXC;

  for (int i = tid; i < MAXC; i += 256)
    ssc[i] = (i < M) ? cand[((size_t)n * MAXC + i) * 6 + 4] : -1.f;
  __syncthreads();

  for (int det = 0; det < NMS_KEEP; ++det) {
    float best = -1.f; int bi = -1;
    for (int i = tid; i < MAXC; i += 256)
      if (ssc[i] > best) { best = ssc[i]; bi = i; }
    rv[tid] = best; ri[tid] = bi;
    __syncthreads();
    for (int off = 128; off > 0; off >>= 1) {
      if (tid < off && rv[tid + off] > rv[tid]) { rv[tid] = rv[tid + off]; ri[tid] = ri[tid + off]; }
      __syncthreads();
    }
    int win = ri[0]; float wscore = rv[0];
    bool valid = (win >= 0) && (wscore > 0.f);
    if (tid == 0) {
      float* ob = out + ((size_t)n * NMS_KEEP + det) * 4;
      if (valid) {
        const float* cp = cand + ((size_t)n * MAXC + win) * 6;
        float offc = cp[5] * 2.f * IMG_SIZE;
        ob[0] = cp[0]; ob[1] = cp[1]; ob[2] = cp[2]; ob[3] = cp[3];
        out[1600 + n * NMS_KEEP + det] = wscore;
        out[2000 + n * NMS_KEEP + det] = cp[5];
        wb[0] = cp[0] + offc; wb[1] = cp[1] + offc;
        wb[2] = cp[2] + offc; wb[3] = cp[3] + offc;
        ssc[win] = -1.f;
      } else {
        ob[0] = ob[1] = ob[2] = ob[3] = 0.f;
        out[1600 + n * NMS_KEEP + det] = 0.f;
        out[2000 + n * NMS_KEEP + det] = -1.f;
      }
    }
    __syncthreads();
    if (valid) {
      float bx1 = wb[0], by1 = wb[1], bx2 = wb[2], by2 = wb[3];
      float barea = (bx2 - bx1) * (by2 - by1);
      for (int i = tid; i < MAXC; i += 256) {
        if (ssc[i] <= 0.f) continue;
        const float* cp = cand + ((size_t)n * MAXC + i) * 6;
        float offc = cp[5] * 2.f * IMG_SIZE;
        float cx1 = cp[0] + offc, cy1 = cp[1] + offc;
        float cx2 = cp[2] + offc, cy2 = cp[3] + offc;
        float ix1 = fmaxf(bx1, cx1), iy1 = fmaxf(by1, cy1);
        float ix2 = fminf(bx2, cx2), iy2 = fminf(by2, cy2);
        float inter = fmaxf(ix2 - ix1, 0.f) * fmaxf(iy2 - iy1, 0.f);
        float iou = inter / (barea + (cx2 - cx1) * (cy2 - cy1) - inter + 1e-9f);
        if (iou > 0.6f) ssc[i] = -1.f;
      }
    }
    __syncthreads();
  }
}

// ---------------------------------------------------------------- launch
extern "C" void kernel_launch(void* const* d_in, const int* in_sizes, int n_in,
                              void* d_out, int out_size, void* d_ws, size_t ws_size,
                              hipStream_t stream) {
  (void)in_sizes; (void)n_in; (void)out_size; (void)ws_size;
  const float* p[5] = {(const float*)d_in[0], (const float*)d_in[1], (const float*)d_in[2],
                       (const float*)d_in[3], (const float*)d_in[4]};
  const float* cls_tw  = (const float*)d_in[5];
  const float* cls_tb  = (const float*)d_in[6];
  const float* bbox_tw = (const float*)d_in[7];
  const float* bbox_tb = (const float*)d_in[8];
  const float* wcls    = (const float*)d_in[9];
  const float* bcls    = (const float*)d_in[10];
  const float* wbox    = (const float*)d_in[11];
  const float* bbox_b  = (const float*)d_in[12];
  const float* wctr    = (const float*)d_in[13];
  const float* bctr    = (const float*)d_in[14];
  const float* scales  = (const float*)d_in[15];

  static const int   HS[5]  = {100, 50, 25, 13, 7};
  static const float STR[5] = {8.f, 16.f, 32.f, 64.f, 128.f};

  size_t off = 0;
  auto carve = [&](size_t bytes) -> void* {
    void* r = (char*)d_ws + off;
    off += (bytes + 255) & ~(size_t)255;
    return r;
  };
  const size_t ACTB = (size_t)NIMG * CIN * 100 * 100 * sizeof(unsigned short);  // bf16 NHWC
  unsigned short* bufIn = (unsigned short*)carve(ACTB);
  unsigned short* buf0  = (unsigned short*)carve(ACTB);
  unsigned short* buf1  = (unsigned short*)carve(ACTB);
  unsigned short* buf2  = (unsigned short*)carve(ACTB);
  float* headout = (float*)carve((size_t)NIMG * 100 * 100 * 96 * sizeof(float));
  unsigned short* wfT = (unsigned short*)carve((size_t)8 * FRAG_PER_CONV * 2);
  unsigned short* wfH = (unsigned short*)carve((size_t)HEAD_FRAG * 2);
  float* bias96 = (float*)carve(96 * sizeof(float));
  float* cand   = (float*)carve((size_t)NIMG * MAXC * 6 * sizeof(float));
  int*   cnt    = (int*)carve(64);

  // one-time (per call) weight conversion: f32 -> bf16 WMMA fragments
  for (int i = 0; i < 4; ++i) {
    prep_tower_w<<<(FRAG_PER_CONV + 255) / 256, 256, 0, stream>>>(
        cls_tw + (size_t)i * CIN * CIN * 9, wfT + (size_t)i * FRAG_PER_CONV);
    prep_tower_w<<<(FRAG_PER_CONV + 255) / 256, 256, 0, stream>>>(
        bbox_tw + (size_t)i * CIN * CIN * 9, wfT + (size_t)(4 + i) * FRAG_PER_CONV);
  }
  prep_head_w<<<(HEAD_FRAG + 255) / 256, 256, 0, stream>>>(wcls, wbox, wctr, wfH);
  prep_bias96<<<1, 96, 0, stream>>>(bcls, bbox_b, bctr, bias96);
  zero_cnt<<<1, 32, 0, stream>>>(cnt);

  for (int l = 0; l < 5; ++l) {
    int H = HS[l], W = HS[l], HW = H * W;
    int wt = (W + 15) / 16;
    int ht = (H + ROWS - 1) / ROWS;
    dim3 cg(wt * ht, NIMG);
    int tot = NIMG * CIN * HW;
    cvt_to_nhwc_bf16<<<(tot + 255) / 256, 256, 0, stream>>>(p[l], bufIn, H, W);
    // cls tower: in -> buf0 -> buf1 -> buf0 -> buf1   (final in buf1)
    conv3x3_wmma<<<cg, 512, 0, stream>>>(bufIn, wfT + 0 * (size_t)FRAG_PER_CONV, cls_tb + 0,   buf0, H, W);
    conv3x3_wmma<<<cg, 512, 0, stream>>>(buf0,  wfT + 1 * (size_t)FRAG_PER_CONV, cls_tb + 256, buf1, H, W);
    conv3x3_wmma<<<cg, 512, 0, stream>>>(buf1,  wfT + 2 * (size_t)FRAG_PER_CONV, cls_tb + 512, buf0, H, W);
    conv3x3_wmma<<<cg, 512, 0, stream>>>(buf0,  wfT + 3 * (size_t)FRAG_PER_CONV, cls_tb + 768, buf1, H, W);
    // bbox tower: in -> buf2 -> buf0 -> buf2 -> buf0  (final in buf0)
    conv3x3_wmma<<<cg, 512, 0, stream>>>(bufIn, wfT + 4 * (size_t)FRAG_PER_CONV, bbox_tb + 0,   buf2, H, W);
    conv3x3_wmma<<<cg, 512, 0, stream>>>(buf2,  wfT + 5 * (size_t)FRAG_PER_CONV, bbox_tb + 256, buf0, H, W);
    conv3x3_wmma<<<cg, 512, 0, stream>>>(buf0,  wfT + 6 * (size_t)FRAG_PER_CONV, bbox_tb + 512, buf2, H, W);
    conv3x3_wmma<<<cg, 512, 0, stream>>>(buf2,  wfT + 7 * (size_t)FRAG_PER_CONV, bbox_tb + 768, buf0, H, W);
    // heads + scoring
    dim3 hg((HW + 15) / 16, NIMG);
    head_wmma<<<hg, 192, 0, stream>>>(buf1, buf0, wfH, bias96, headout, HW);
    int stot = NIMG * HW;
    score_decode<<<(stot + 255) / 256, 256, 0, stream>>>(headout, scales, l, H, W, STR[l], cand, cnt);
  }
  nms_kernel<<<NIMG, 256, 0, stream>>>(cand, cnt, (float*)d_out);
}
